// Hie_Couple_6485400617374
// MI455X (gfx1250) — matrix operand order repaired and checked by reading, not compiled, not run on hardware
//
#include <hip/hip_runtime.h>

// ---------------------------------------------------------------------------
// CDNA5 (gfx1250) implementation of Hie_Couple forward (dead code removed).
//   Q=xs@wqs^T+bqs, K=xf@wkf^T+bkf, Vs=xs@wvs^T+bvs
//   QF=xf@cfa_wq^T+cfa_bq, KF=xs_fine@cfa_wk^T+cfa_bk
//   C_M  = Q@K^T/16            (output #3)
//   C_MT = K@Q^T/16            (transposed view for axis=-2 entmax)
//   ZF   = QF@KF^T/16 ; C_M_F = entmax15_rows(ZF)         [8,1024,4096]
//   P    = C_M_F @ S_s                                    [8,1024(Nf),1024(Ns)]
//   GT   = C_MT + log(clip(P,1e-6)) ; A_fT = entmax15_rows(GT)
//   xf_  = (A_fT @ Vs) @ wof^T + bof   (output #2)
//   out1 = xf_ + LayerNorm(xf; fn_g, fn_b)  (output #1)
// All GEMM operands are pre-converted to bf16 exactly once (inputs via cvt
// kernels, intermediates written bf16 by GEMM/entmax epilogues), so the
// v_wmma_f32_16x16x32_bf16 inner loop has zero conversion VALU and half the
// global read traffic. entmax15 solved exactly (to f32 precision) by 32-step
// bisection on f(tau)=sum((z-tau)+^2)=1, tau in [-1,0].
// ---------------------------------------------------------------------------

typedef __attribute__((ext_vector_type(16))) __bf16 v16bf;
typedef __attribute__((ext_vector_type(8)))  float  v8f;

__device__ __forceinline__ unsigned short f32_to_bf16(float f) {
  union { float f; unsigned int u; } x; x.f = f;
  unsigned int r = x.u + 0x7FFFu + ((x.u >> 16) & 1u);  // round-nearest-even
  return (unsigned short)(r >> 16);
}

// packed pair conversion: native v_cvt_pk_bf16_f32 when available
__device__ __forceinline__ unsigned pk2_bf16(float a, float b) {
#if __has_builtin(__builtin_amdgcn_cvt_pk_bf16_f32)
  auto r = __builtin_amdgcn_cvt_pk_bf16_f32(a, b);
  unsigned u; __builtin_memcpy(&u, &r, sizeof(u));
  return u;
#else
  return (unsigned)f32_to_bf16(a) | ((unsigned)f32_to_bf16(b) << 16);
#endif
}

union Frag { uint4 u[2]; v16bf v; };

// ---------------------------------------------------------------------------
// GEMM: C[M,N] = alpha * A[M,K] * op(B) (+ bias[col]), batched over grid.z.
// A, B are bf16 (ushort); C is f32 or bf16 (OUT_BF16).
//   B_NT=true : B stored [N,K] (computes A @ B^T)
//   B_NT=false: B stored [K,N] (computes A @ B)
// Workgroup tile 128x64, 8 waves in 4x2; each wave owns 32x32 (4 WMMAs/step).
// Double-buffered LDS; global loads for step k+1 issued before WMMAs of step
// k; tile k+2 prefetched. Requires M%128==0, N%64==0, K%32==0.
// ---------------------------------------------------------------------------
template<bool B_NT, bool OUT_BF16>
__global__ __launch_bounds__(256)
void gemm_bf16(const unsigned short* __restrict__ A,
               const unsigned short* __restrict__ Bm,
               const float* __restrict__ bias, void* __restrict__ Cp,
               int M, int N, int K,
               long long strideA, long long strideB, long long strideC,
               float alpha)
{
  // padded rows: 40 bf16 = 80 bytes (16B-aligned sub-chunks at 0/16/32/48)
  __shared__ unsigned short sA[2][128 * 40];
  __shared__ unsigned short sB[2][64 * 40];

  const int bz = blockIdx.z;
  A += (long long)bz * strideA;
  Bm += (long long)bz * strideB;

  const int m0 = blockIdx.y * 128;
  const int n0 = blockIdx.x * 64;
  const int t = threadIdx.x;
  const int lane = t & 31;
  const int wid = t >> 5;
  const int wm = wid & 3;    // 32-row block within tile
  const int wn = wid >> 2;   // 32-col block within tile

  v8f acc00 = {}, acc01 = {}, acc10 = {}, acc11 = {};

  // A tile 128x32 bf16: 2 chunks of 8 elems per thread
  const int ar0 = t >> 2;              // 0..63
  const int ar1 = ar0 + 64;            // 64..127
  const int ac = (t & 3) * 8;          // 0,8,16,24
  // B tile (NT): 64x32, 1 chunk per thread
  const int br = t >> 2;               // 0..63
  // B tile (NN): 32(K) x 64(N)
  const int kr = t >> 3;               // 0..31
  const int nc = (t & 7) * 8;          // 0..56

  uint4 ra0, ra1, rb;

  // ---- prologue: tile 0 ----
  ra0 = *(const uint4*)(A + (long long)(m0 + ar0) * K + ac);
  ra1 = *(const uint4*)(A + (long long)(m0 + ar1) * K + ac);
  if (B_NT) rb = *(const uint4*)(Bm + (long long)(n0 + br) * K + ac);
  else      rb = *(const uint4*)(Bm + (long long)kr * N + n0 + nc);
  *(uint4*)(&sA[0][ar0 * 40 + ac]) = ra0;
  *(uint4*)(&sA[0][ar1 * 40 + ac]) = ra1;
  if (B_NT) {
    *(uint4*)(&sB[0][br * 40 + ac]) = rb;
  } else {
    union { uint4 v; unsigned short h[8]; } u; u.v = rb;
#pragma unroll
    for (int i = 0; i < 8; ++i) sB[0][(nc + i) * 40 + kr] = u.h[i];
  }
  __syncthreads();

  int cur = 0;
  for (int kk = 0; kk < K; kk += 32) {
    const bool has_next = (kk + 32) < K;

    // ---- issue global loads for tile k+1 (overlap with WMMAs below) ----
    if (has_next) {
      const int kn = kk + 32;
      ra0 = *(const uint4*)(A + (long long)(m0 + ar0) * K + kn + ac);
      ra1 = *(const uint4*)(A + (long long)(m0 + ar1) * K + kn + ac);
      if (B_NT) rb = *(const uint4*)(Bm + (long long)(n0 + br) * K + kn + ac);
      else      rb = *(const uint4*)(Bm + (long long)(kn + kr) * N + n0 + nc);
      // prefetch tile k+2 toward L2 (global_prefetch_b8)
      if (kk + 64 < K) {
        __builtin_prefetch(A + (long long)(m0 + ar0) * K + kk + 64 + ac, 0, 1);
        if (B_NT)
          __builtin_prefetch(Bm + (long long)(n0 + br) * K + kk + 64 + ac, 0, 1);
        else
          __builtin_prefetch(Bm + (long long)(kk + 64 + kr) * N + n0 + nc, 0, 1);
      }
    }

    // ---- fragments per ISA 7.12.2 layouts ----
    Frag fa0, fa1, fb0, fb1;
    {
      // A (16x32): lanes0-15 M=lane,K0..7&16..23; lanes16-31 M=lane-16,K8..15&24..31
      const int arow = wm * 32 + (lane & 15);
      const int ak = (lane >> 4) * 8;  // 0 or 8
      fa0.u[0] = *(const uint4*)(&sA[cur][arow * 40 + ak]);
      fa0.u[1] = *(const uint4*)(&sA[cur][arow * 40 + ak + 16]);
      fa1.u[0] = *(const uint4*)(&sA[cur][(arow + 16) * 40 + ak]);
      fa1.u[1] = *(const uint4*)(&sA[cur][(arow + 16) * 40 + ak + 16]);
      // B (32x16): lanes0-15 N=lane,K0..7&16..23; lanes16-31 N=lane-16,K8..15&24..31
      const int bcol = wn * 32 + (lane & 15);
      const int bk = (lane >> 4) * 8;  // 0 or 8
      fb0.u[0] = *(const uint4*)(&sB[cur][bcol * 40 + bk]);
      fb0.u[1] = *(const uint4*)(&sB[cur][bcol * 40 + bk + 16]);
      fb1.u[0] = *(const uint4*)(&sB[cur][(bcol + 16) * 40 + bk]);
      fb1.u[1] = *(const uint4*)(&sB[cur][(bcol + 16) * 40 + bk + 16]);
    }

    acc00 = __builtin_amdgcn_wmma_f32_16x16x32_bf16(false, fa0.v, false, fb0.v,
                                                    (short)0, acc00, false, false);
    acc01 = __builtin_amdgcn_wmma_f32_16x16x32_bf16(false, fa0.v, false, fb1.v,
                                                    (short)0, acc01, false, false);
    acc10 = __builtin_amdgcn_wmma_f32_16x16x32_bf16(false, fa1.v, false, fb0.v,
                                                    (short)0, acc10, false, false);
    acc11 = __builtin_amdgcn_wmma_f32_16x16x32_bf16(false, fa1.v, false, fb1.v,
                                                    (short)0, acc11, false, false);

    // ---- store tile k+1 into the other buffer (no conversion needed) ----
    if (has_next) {
      const int nxt = cur ^ 1;
      *(uint4*)(&sA[nxt][ar0 * 40 + ac]) = ra0;
      *(uint4*)(&sA[nxt][ar1 * 40 + ac]) = ra1;
      if (B_NT) {
        *(uint4*)(&sB[nxt][br * 40 + ac]) = rb;
      } else {
        union { uint4 v; unsigned short h[8]; } u; u.v = rb;
#pragma unroll
        for (int i = 0; i < 8; ++i) sB[nxt][(nc + i) * 40 + kr] = u.h[i];
      }
    }
    __syncthreads();
    cur ^= 1;
  }

  // ---- epilogue: C/D layout — VGPR i: lanes0-15 (M=i,N=lane), lanes16-31 (M=8+i) ----
  const int er = m0 + wm * 32 + ((lane < 16) ? 0 : 8);
  const int ec = n0 + wn * 32 + (lane & 15);
  float bb0 = 0.f, bb1 = 0.f;
  if (bias) { bb0 = bias[ec]; bb1 = bias[ec + 16]; }
  if constexpr (OUT_BF16) {
    unsigned short* C = (unsigned short*)Cp + (long long)blockIdx.z * strideC;
#pragma unroll
    for (int i = 0; i < 8; ++i) {
      C[(long long)(er + i) * N + ec]           = f32_to_bf16(alpha * acc00[i] + bb0);
      C[(long long)(er + i) * N + ec + 16]      = f32_to_bf16(alpha * acc01[i] + bb1);
      C[(long long)(er + 16 + i) * N + ec]      = f32_to_bf16(alpha * acc10[i] + bb0);
      C[(long long)(er + 16 + i) * N + ec + 16] = f32_to_bf16(alpha * acc11[i] + bb1);
    }
  } else {
    float* C = (float*)Cp + (long long)blockIdx.z * strideC;
#pragma unroll
    for (int i = 0; i < 8; ++i) {
      C[(long long)(er + i) * N + ec]           = alpha * acc00[i] + bb0;
      C[(long long)(er + i) * N + ec + 16]      = alpha * acc01[i] + bb1;
      C[(long long)(er + 16 + i) * N + ec]      = alpha * acc10[i] + bb0;
      C[(long long)(er + 16 + i) * N + ec + 16] = alpha * acc11[i] + bb1;
    }
  }
}

// ---------------------------------------------------------------------------
// f32 -> bf16 bulk conversion (8 elems/thread/iter, packed pair cvt)
// ---------------------------------------------------------------------------
__global__ __launch_bounds__(256)
void cvt_f32_bf16_kernel(const float* __restrict__ src,
                         unsigned short* __restrict__ dst, long long n8)
{
  long long i = (long long)blockIdx.x * 256 + threadIdx.x;
  const long long stride = (long long)gridDim.x * 256;
  for (; i < n8; i += stride) {
    float4 f0 = ((const float4*)src)[2 * i];
    float4 f1 = ((const float4*)src)[2 * i + 1];
    uint4 r;
    r.x = pk2_bf16(f0.x, f0.y);
    r.y = pk2_bf16(f0.z, f0.w);
    r.z = pk2_bf16(f1.x, f1.y);
    r.w = pk2_bf16(f1.z, f1.w);
    ((uint4*)dst)[i] = r;
  }
}

// ---------------------------------------------------------------------------
// Reductions (block = 256 threads = 8 wave32)
// ---------------------------------------------------------------------------
__device__ __forceinline__ float wave_sum32(float v) {
#pragma unroll
  for (int off = 16; off; off >>= 1) v += __shfl_xor(v, off, 32);
  return v;
}
__device__ __forceinline__ float wave_max32(float v) {
#pragma unroll
  for (int off = 16; off; off >>= 1) v = fmaxf(v, __shfl_xor(v, off, 32));
  return v;
}
__device__ __forceinline__ float block_sum(float v, float* red) {
  v = wave_sum32(v);
  __syncthreads();
  if ((threadIdx.x & 31) == 0) red[threadIdx.x >> 5] = v;
  __syncthreads();
  float r = 0.f;
#pragma unroll
  for (int i = 0; i < 8; ++i) r += red[i];
  return r;
}
__device__ __forceinline__ float block_max(float v, float* red) {
  v = wave_max32(v);
  __syncthreads();
  if ((threadIdx.x & 31) == 0) red[threadIdx.x >> 5] = v;
  __syncthreads();
  float r = red[0];
#pragma unroll
  for (int i = 1; i < 8; ++i) r = fmaxf(r, red[i]);
  return r;
}

// ---------------------------------------------------------------------------
// entmax-1.5 over contiguous rows of length L = NV4*1024; reads f32 scores,
// writes bf16 probabilities (consumed only by bf16 GEMMs).
// z=(z-max)*0.5; tau = root of sum((z-tau)+^2)=1 via 32-step bisection on
// [-1,0]; p=(z-tau)+^2. One block per row; row lives in registers.
// ---------------------------------------------------------------------------
template<int NV4>
__global__ __launch_bounds__(256)
void entmax15_rows_kernel(const float* __restrict__ Z,
                          unsigned short* __restrict__ Pout)
{
  __shared__ float red[8];
  const int t = threadIdx.x;
  const float4* row = (const float4*)(Z + (long long)blockIdx.x * (NV4 * 1024));
  float4 zl[NV4];
#pragma unroll
  for (int i = 0; i < NV4; ++i) zl[i] = row[t + i * 256];

  float mx = -3.402823466e+38f;
#pragma unroll
  for (int i = 0; i < NV4; ++i) {
    float* f = (float*)&zl[i];
#pragma unroll
    for (int j = 0; j < 4; ++j) mx = fmaxf(mx, f[j]);
  }
  mx = block_max(mx, red);

#pragma unroll
  for (int i = 0; i < NV4; ++i) {
    float* f = (float*)&zl[i];
#pragma unroll
    for (int j = 0; j < 4; ++j) f[j] = (f[j] - mx) * 0.5f;
  }

  float lo = -1.0f, hi = 0.0f;
  for (int it = 0; it < 32; ++it) {
    const float mid = 0.5f * (lo + hi);
    float s = 0.0f;
#pragma unroll
    for (int i = 0; i < NV4; ++i) {
      float* f = (float*)&zl[i];
#pragma unroll
      for (int j = 0; j < 4; ++j) {
        const float d = fmaxf(f[j] - mid, 0.0f);
        s += d * d;
      }
    }
    s = block_sum(s, red);
    if (s >= 1.0f) lo = mid; else hi = mid;   // uniform across block
  }
  const float tau = 0.5f * (lo + hi);

  uint2* orow = (uint2*)(Pout + (long long)blockIdx.x * (NV4 * 1024));
#pragma unroll
  for (int i = 0; i < NV4; ++i) {
    float* f = (float*)&zl[i];
    float p[4];
#pragma unroll
    for (int j = 0; j < 4; ++j) {
      const float d = fmaxf(f[j] - tau, 0.0f);
      p[j] = d * d;
    }
    uint2 o;
    o.x = pk2_bf16(p[0], p[1]);
    o.y = pk2_bf16(p[2], p[3]);
    orow[t + i * 256] = o;
  }
}

// P[i] = CMT[i] + log(max(P[i], 1e-6))
__global__ __launch_bounds__(256)
void logadd_kernel(const float* __restrict__ cmt, float* __restrict__ p, long long n)
{
  long long i = (long long)blockIdx.x * 256 + threadIdx.x;
  const long long stride = (long long)gridDim.x * 256;
  for (; i < n; i += stride) p[i] = cmt[i] + logf(fmaxf(p[i], 1e-6f));
}

// out[row,:] = xfp[row,:] + LayerNorm(xf[row,:]; g, b)   (D=256, one block/row)
__global__ __launch_bounds__(256)
void ln_add_kernel(const float* __restrict__ xf, const float* __restrict__ xfp,
                   const float* __restrict__ g, const float* __restrict__ bb,
                   float* __restrict__ out)
{
  __shared__ float red[8];
  const long long base = (long long)blockIdx.x * 256;
  const int t = threadIdx.x;
  const float x = xf[base + t];
  const float mu = block_sum(x, red) * (1.0f / 256.0f);
  const float d = x - mu;
  const float var = block_sum(d * d, red) * (1.0f / 256.0f);
  const float y = d * rsqrtf(var + 1e-5f) * g[t] + bb[t];
  out[base + t] = xfp[base + t] + y;
}

// ---------------------------------------------------------------------------
extern "C" void kernel_launch(void* const* d_in, const int* in_sizes, int n_in,
                              void* d_out, int out_size, void* d_ws, size_t ws_size,
                              hipStream_t stream)
{
  (void)in_sizes; (void)n_in; (void)out_size; (void)ws_size;

  const float* xs      = (const float*)d_in[0];
  const float* xf      = (const float*)d_in[1];
  const float* xs_fine = (const float*)d_in[2];
  const float* S_s     = (const float*)d_in[4];
  const float* cfa_wq  = (const float*)d_in[6];
  const float* cfa_bq  = (const float*)d_in[7];
  const float* cfa_wk  = (const float*)d_in[8];
  const float* cfa_bk  = (const float*)d_in[9];
  const float* wqs     = (const float*)d_in[14];
  const float* bqs     = (const float*)d_in[15];
  const float* wkf     = (const float*)d_in[16];
  const float* bkf     = (const float*)d_in[17];
  const float* wvs     = (const float*)d_in[18];
  const float* bvs     = (const float*)d_in[19];
  const float* wof     = (const float*)d_in[24];
  const float* bof     = (const float*)d_in[25];
  const float* fn_g    = (const float*)d_in[28];
  const float* fn_b    = (const float*)d_in[29];
  // dead: xf_fine, S_f, csa_*, wvf/bvf, wos/bos, sn_g/sn_b

  float* out = (float*)d_out;

  const long long ND  = 1024LL * 256;     // per-batch [1024,256]
  const long long NFD = 4096LL * 256;     // per-batch [4096,256]
  const long long NN_ = 1024LL * 1024;    // per-batch [1024,1024]
  const long long NZ  = 1024LL * 4096;    // per-batch [1024,4096]
  const long long WW  = 256LL * 256;      // weight [256,256]

  // f32 scratch
  float* ZF  = (float*)d_ws;      // [8,1024,4096]  scores (f32)
  float* Pb  = ZF + 8 * NZ;       // [8,1024,1024]  P -> GT (f32)
  float* CMT = Pb + 8 * NN_;      // [8,1024,1024]  C_M^T (f32)
  // bf16 scratch
  unsigned short* h = (unsigned short*)(CMT + 8 * NN_);
  unsigned short* xs_h  = h; h += 8 * ND;
  unsigned short* xf_h  = h; h += 8 * ND;
  unsigned short* xsf_h = h; h += 8 * NFD;
  unsigned short* Ss_h  = h; h += 8 * NZ;
  unsigned short* wqs_h = h; h += WW;
  unsigned short* wkf_h = h; h += WW;
  unsigned short* wvs_h = h; h += WW;
  unsigned short* cwq_h = h; h += WW;
  unsigned short* cwk_h = h; h += WW;
  unsigned short* wof_h = h; h += WW;
  unsigned short* Qh    = h; h += 8 * ND;
  unsigned short* Kh    = h; h += 8 * ND;
  unsigned short* Vsh   = h; h += 8 * ND;
  unsigned short* QFh   = h; h += 8 * ND;
  unsigned short* KFh   = h; h += 8 * NFD;
  unsigned short* ZFh   = h; h += 8 * NZ;     // C_M_F (bf16)
  unsigned short* AfTh  = h; h += 8 * NN_;    // A_f^T (bf16)
  unsigned short* XFmh  = h; h += 8 * ND;     // A_f^T @ Vs (bf16)

  float* out_xs = out;                    // [8,1024,256]
  float* out_xf = out + 8 * ND;           // [8,1024,256]
  float* out_cm = out + 16 * ND;          // [8,1024,1024]

  const dim3 blk(256);
  const float inv_sqrt_d = 0.0625f;       // 1/sqrt(256)

  // ---- one-time f32 -> bf16 conversions of GEMM operands ----
  cvt_f32_bf16_kernel<<<1024, blk, 0, stream>>>(xs, xs_h, 8 * ND / 8);
  cvt_f32_bf16_kernel<<<1024, blk, 0, stream>>>(xf, xf_h, 8 * ND / 8);
  cvt_f32_bf16_kernel<<<2048, blk, 0, stream>>>(xs_fine, xsf_h, 8 * NFD / 8);
  cvt_f32_bf16_kernel<<<4096, blk, 0, stream>>>(S_s, Ss_h, 8 * NZ / 8);
  cvt_f32_bf16_kernel<<<32, blk, 0, stream>>>(wqs, wqs_h, WW / 8);
  cvt_f32_bf16_kernel<<<32, blk, 0, stream>>>(wkf, wkf_h, WW / 8);
  cvt_f32_bf16_kernel<<<32, blk, 0, stream>>>(wvs, wvs_h, WW / 8);
  cvt_f32_bf16_kernel<<<32, blk, 0, stream>>>(cfa_wq, cwq_h, WW / 8);
  cvt_f32_bf16_kernel<<<32, blk, 0, stream>>>(cfa_wk, cwk_h, WW / 8);
  cvt_f32_bf16_kernel<<<32, blk, 0, stream>>>(wof, wof_h, WW / 8);

  // ---- linears (bf16 out; consumed only by GEMMs). Grid: (N/64, M/128, b) ----
  gemm_bf16<true, true><<<dim3(4, 8, 8), blk, 0, stream>>>(xs_h, wqs_h, bqs, Qh, 1024, 256, 256, ND, 0, ND, 1.0f);
  gemm_bf16<true, true><<<dim3(4, 8, 8), blk, 0, stream>>>(xf_h, wkf_h, bkf, Kh, 1024, 256, 256, ND, 0, ND, 1.0f);
  gemm_bf16<true, true><<<dim3(4, 8, 8), blk, 0, stream>>>(xs_h, wvs_h, bvs, Vsh, 1024, 256, 256, ND, 0, ND, 1.0f);
  gemm_bf16<true, true><<<dim3(4, 8, 8), blk, 0, stream>>>(xf_h, cwq_h, cfa_bq, QFh, 1024, 256, 256, ND, 0, ND, 1.0f);
  gemm_bf16<true, true><<<dim3(4, 32, 8), blk, 0, stream>>>(xsf_h, cwk_h, cfa_bk, KFh, 4096, 256, 256, NFD, 0, NFD, 1.0f);

  // ---- C_M (output #3, f32) and C_M^T ----
  gemm_bf16<true, false><<<dim3(16, 8, 8), blk, 0, stream>>>(Qh, Kh, nullptr, out_cm, 1024, 1024, 256, ND, ND, NN_, inv_sqrt_d);
  gemm_bf16<true, false><<<dim3(16, 8, 8), blk, 0, stream>>>(Kh, Qh, nullptr, CMT, 1024, 1024, 256, ND, ND, NN_, inv_sqrt_d);

  // ---- ZF = QF @ KF^T / 16 (f32) ; C_M_F = entmax15 rows -> bf16 ----
  gemm_bf16<true, false><<<dim3(64, 8, 8), blk, 0, stream>>>(QFh, KFh, nullptr, ZF, 1024, 4096, 256, ND, NFD, NZ, inv_sqrt_d);
  entmax15_rows_kernel<4><<<8 * 1024, blk, 0, stream>>>(ZF, ZFh);

  // ---- P = C_M_F @ S_s (NN, K=4096, f32 out) ----
  gemm_bf16<false, false><<<dim3(16, 8, 8), blk, 0, stream>>>(ZFh, Ss_h, nullptr, Pb, 1024, 1024, 4096, NZ, NZ, NN_, 1.0f);

  // ---- GT = C_M^T + log(clip(P)) ; A_fT = entmax15 rows -> bf16 ----
  logadd_kernel<<<4096, blk, 0, stream>>>(CMT, Pb, 8 * NN_);
  entmax15_rows_kernel<1><<<8 * 1024, blk, 0, stream>>>(Pb, AfTh);

  // ---- xf_ = (A_fT @ Vs) @ wof^T + bof (output #2) ----
  gemm_bf16<false, true><<<dim3(4, 8, 8), blk, 0, stream>>>(AfTh, Vsh, nullptr, XFmh, 1024, 256, 1024, NN_, ND, ND, 1.0f);
  gemm_bf16<true, false><<<dim3(4, 8, 8), blk, 0, stream>>>(XFmh, wof_h, bof, out_xf, 1024, 256, 256, ND, 0, ND, 1.0f);

  // ---- out #1 = xf_ + LayerNorm(xf) ----
  ln_add_kernel<<<8 * 1024, blk, 0, stream>>>(xf, out_xf, fn_g, fn_b, out_xs);
}
